// AttentiveStatistiC_15814069584292
// MI455X (gfx1250) — matrix operand order repaired and checked by reading, not compiled, not run on hardware
//
#include <hip/hip_runtime.h>
#include <hip/hip_bf16.h>
#include <math.h>

// ---------------------------------------------------------------------------
// Problem constants (from reference: B=32, T=512, HID=256, 4x4 spatial)
// ---------------------------------------------------------------------------
#define BB    32
#define TT    512
#define HIDC  256
#define SP    16          // 4*4 spatial positions
#define CP    (HIDC*SP)   // 4096 = flattened (C,4,4)
#define NHEAD 4
#define DH    (HIDC/NHEAD) // 64
#define FEAT  CP          // 4096
#define CDIM  128
#define KSPLIT 8

typedef __attribute__((ext_vector_type(16))) __bf16 v16bf;
typedef __attribute__((ext_vector_type(8)))  float  v8f;

__device__ __forceinline__ void f4cvt(v16bf& d, int base, float4 f) {
    d[base + 0] = (__bf16)f.x;
    d[base + 1] = (__bf16)f.y;
    d[base + 2] = (__bf16)f.z;
    d[base + 3] = (__bf16)f.w;
}

// ---------------------------------------------------------------------------
// 1) z[b,cp] = mean_t h[b,t,cp]          (one streaming pass over h)
// ---------------------------------------------------------------------------
__global__ __launch_bounds__(256) void mean_kernel(const float* __restrict__ h,
                                                   float* __restrict__ z) {
    int idx = blockIdx.x * 256 + threadIdx.x;      // B*CP threads
    int b = idx / CP, cp = idx % CP;
    const float* p = h + (size_t)b * TT * CP + cp;
    float acc = 0.f;
    #pragma unroll 4
    for (int t = 0; t < TT; ++t) acc += p[(size_t)t * CP];
    z[idx] = acc * (1.0f / TT);
}

// ---------------------------------------------------------------------------
// 2) q[b,o,p] = bq[o] + sum_c Wq[o,c] * z[b,c,p]
// ---------------------------------------------------------------------------
__global__ __launch_bounds__(256) void qproj_kernel(const float* __restrict__ z,
                                                    const float* __restrict__ Wq,
                                                    const float* __restrict__ bq,
                                                    float* __restrict__ q) {
    int idx = blockIdx.x * 256 + threadIdx.x;      // B*CP threads
    int b = idx / CP, r = idx % CP, o = r / SP, p = r % SP;
    const float* zr = z + (size_t)b * CP + p;
    const float* wr = Wq + (size_t)o * HIDC;
    float acc = bq[o];
    #pragma unroll 4
    for (int c = 0; c < HIDC; ++c) acc += wr[c] * zr[c * SP];
    q[idx] = acc;
}

// ---------------------------------------------------------------------------
// 3) qk[b,n,c,p] = sum_{j<DH} q[b, n*DH+j, p] * Wk[n*DH+j, c]
// ---------------------------------------------------------------------------
__global__ __launch_bounds__(256) void qk_kernel(const float* __restrict__ q,
                                                 const float* __restrict__ Wk,
                                                 float* __restrict__ qk) {
    int idx = blockIdx.x * 256 + threadIdx.x;      // B*NHEAD*CP threads
    int b = idx / (NHEAD * CP);
    int r = idx % (NHEAD * CP);
    int n = r / CP, s = r % CP, c = s / SP, p = s % SP;
    float acc = 0.f;
    #pragma unroll 4
    for (int j = 0; j < DH; ++j) {
        int o = n * DH + j;
        acc += q[(size_t)b * CP + o * SP + p] * Wk[(size_t)o * HIDC + c];
    }
    qk[idx] = acc;
}

// qb[b,n] = sum_{o in head n, p} q[b,o,p] * bk[o]
__global__ void qb_kernel(const float* __restrict__ q,
                          const float* __restrict__ bk,
                          float* __restrict__ qb) {
    int idx = threadIdx.x;                          // B*NHEAD = 128 threads
    if (idx >= BB * NHEAD) return;
    int b = idx / NHEAD, n = idx % NHEAD;
    float acc = 0.f;
    for (int j = 0; j < DH; ++j) {
        int o = n * DH + j;
        float bko = bk[o];
        for (int p = 0; p < SP; ++p)
            acc += q[(size_t)b * CP + o * SP + p] * bko;
    }
    qb[idx] = acc;
}

// ---------------------------------------------------------------------------
// 4) sim[b,n,t] = (<qk[b,n,:], h[b,t,:]> + qb[b,n]) * 1/32
//    one wave per (b,t); second streaming pass over h
// ---------------------------------------------------------------------------
__global__ __launch_bounds__(256) void sim_kernel(const float* __restrict__ h,
                                                  const float* __restrict__ qk,
                                                  const float* __restrict__ qb,
                                                  float* __restrict__ sim) {
    int w = blockIdx.x * 8 + (threadIdx.x >> 5);   // global wave id = b*T + t
    int lane = threadIdx.x & 31;
    int b = w / TT, t = w % TT;
    const float* hp  = h  + ((size_t)b * TT + t) * CP;
    const float* qkp = qk + (size_t)b * NHEAD * CP;
    float a0 = 0.f, a1 = 0.f, a2 = 0.f, a3 = 0.f;
    for (int cp = lane; cp < CP; cp += 32) {
        float hv = hp[cp];
        a0 += hv * qkp[0 * CP + cp];
        a1 += hv * qkp[1 * CP + cp];
        a2 += hv * qkp[2 * CP + cp];
        a3 += hv * qkp[3 * CP + cp];
    }
    #pragma unroll
    for (int off = 16; off > 0; off >>= 1) {
        a0 += __shfl_xor(a0, off, 32);
        a1 += __shfl_xor(a1, off, 32);
        a2 += __shfl_xor(a2, off, 32);
        a3 += __shfl_xor(a3, off, 32);
    }
    if (lane == 0) {
        const float scale = 0.03125f;              // 1/sqrt(4*4*64)
        sim[((size_t)b * NHEAD + 0) * TT + t] = (a0 + qb[b * NHEAD + 0]) * scale;
        sim[((size_t)b * NHEAD + 1) * TT + t] = (a1 + qb[b * NHEAD + 1]) * scale;
        sim[((size_t)b * NHEAD + 2) * TT + t] = (a2 + qb[b * NHEAD + 2]) * scale;
        sim[((size_t)b * NHEAD + 3) * TT + t] = (a3 + qb[b * NHEAD + 3]) * scale;
    }
}

// ---------------------------------------------------------------------------
// 5) softmax over t per (b,n); one wave per head, block per b
// ---------------------------------------------------------------------------
__global__ __launch_bounds__(128) void softmax_kernel(const float* __restrict__ sim,
                                                      float* __restrict__ att_ws,
                                                      float* __restrict__ att_out) {
    int b = blockIdx.x;
    int n = threadIdx.x >> 5, lane = threadIdx.x & 31;
    const float* sp = sim + ((size_t)b * NHEAD + n) * TT;
    float v[16];
    float mx = -3.4e38f;
    #pragma unroll
    for (int i = 0; i < 16; ++i) { v[i] = sp[i * 32 + lane]; mx = fmaxf(mx, v[i]); }
    #pragma unroll
    for (int off = 16; off > 0; off >>= 1) mx = fmaxf(mx, __shfl_xor(mx, off, 32));
    float sum = 0.f;
    #pragma unroll
    for (int i = 0; i < 16; ++i) { v[i] = __expf(v[i] - mx); sum += v[i]; }
    #pragma unroll
    for (int off = 16; off > 0; off >>= 1) sum += __shfl_xor(sum, off, 32);
    float inv = 1.0f / sum;
    size_t base = ((size_t)b * NHEAD + n) * TT;
    #pragma unroll
    for (int i = 0; i < 16; ++i) {
        float a = v[i] * inv;
        att_ws[base + i * 32 + lane]  = a;
        att_out[base + i * 32 + lane] = a;
    }
}

// ---------------------------------------------------------------------------
// 6) hbar[b,n,cp] = sum_t att[b,n,t] * h[b,t,cp]   (third pass over h)
//    att staged into LDS with CDNA5 async copy (GLOBAL_LOAD_ASYNC_TO_LDS_B128,
//    tracked by ASYNCcnt, drained with s_wait_asynccnt before the barrier).
//    16 blocks per b, 256 cp per block.
// ---------------------------------------------------------------------------
__global__ __launch_bounds__(256) void hbar_kernel(const float* __restrict__ h,
                                                   const float* __restrict__ att,
                                                   float* __restrict__ hbar) {
    __shared__ float s_att[NHEAD * TT];            // 8 KB
    int b = blockIdx.x / 16, chunk = blockIdx.x % 16;
    int tid = threadIdx.x;

    // 8192 bytes = 512 x B128 transfers, 2 per thread, global -> LDS direct
    const float* gsrc = att + (size_t)b * NHEAD * TT;
    #pragma unroll
    for (int i = 0; i < 2; ++i) {
        int f0 = (tid + i * 256) * 4;              // float index, 4 floats / op
        unsigned lds_addr = (unsigned)(uintptr_t)(&s_att[f0]);
        unsigned long long gaddr = (unsigned long long)(uintptr_t)(gsrc + f0);
        asm volatile("global_load_async_to_lds_b128 %0, %1, off"
                     :: "v"(lds_addr), "v"(gaddr) : "memory");
    }
    asm volatile("s_wait_asynccnt 0" ::: "memory");
    __syncthreads();

    int cp = chunk * 256 + tid;
    const float* hp = h + (size_t)b * TT * CP + cp;
    float a0 = 0.f, a1 = 0.f, a2 = 0.f, a3 = 0.f;
    for (int t = 0; t < TT; ++t) {
        float hv = hp[(size_t)t * CP];
        a0 += s_att[0 * TT + t] * hv;
        a1 += s_att[1 * TT + t] * hv;
        a2 += s_att[2 * TT + t] * hv;
        a3 += s_att[3 * TT + t] * hv;
    }
    size_t base = (size_t)b * NHEAD * CP + cp;
    hbar[base + 0 * CP] = a0;
    hbar[base + 1 * CP] = a1;
    hbar[base + 2 * CP] = a2;
    hbar[base + 3 * CP] = a3;
}

// ---------------------------------------------------------------------------
// 7) a[b,o,p] = bv[o] + sum_c Wv[o,c] * hbar[b, o/DH, c*SP+p]
// ---------------------------------------------------------------------------
__global__ __launch_bounds__(256) void aproj_kernel(const float* __restrict__ hbar,
                                                    const float* __restrict__ Wv,
                                                    const float* __restrict__ bv,
                                                    float* __restrict__ a_out) {
    int idx = blockIdx.x * 256 + threadIdx.x;      // B*CP threads
    int b = idx / CP, r = idx % CP, o = r / SP, p = r % SP;
    int n = o / DH;
    const float* hb = hbar + ((size_t)b * NHEAD + n) * CP + p;
    const float* wr = Wv + (size_t)o * HIDC;
    float acc = bv[o];
    #pragma unroll 4
    for (int c = 0; c < HIDC; ++c) acc += wr[c] * hb[c * SP];
    a_out[idx] = acc;
}

// ---------------------------------------------------------------------------
// 8) PostPool GEMM:  ACCpart[s][m,n] = X[m, k0:k1] . W[n, k0:k1]
//    bf16 WMMA (f32 accumulate), one wave per (n-tile, k-slice),
//    both 16-row M tiles per wave.  CDNA5 §7.12.2 lane layouts.
// ---------------------------------------------------------------------------
__global__ __launch_bounds__(32) void pp_gemm(const float* __restrict__ X,   // (32,K)
                                              const float* __restrict__ W,   // (N,K)
                                              float* __restrict__ ACC,       // (KSPLIT,32,N)
                                              int N, int K, int kchunk) {
    int lane = threadIdx.x;
    int n0 = blockIdx.x * 16;
    int ks0 = blockIdx.y * kchunk;

    // B-matrix (32x16 of W^T): col = n0 + (lane&15); lanes>=16 hold K-halves 8..15/24..31
    int col = n0 + (lane & 15);
    int kbB = (lane < 16) ? 0 : 8;
    const float* Wp = W + (size_t)col * K + ks0 + kbB;
    // A-matrix (16x32 of X): row = lane&15; lanes>=16 hold K-halves 8..15/24..31
    int rowA = lane & 15;
    int kbA = (lane < 16) ? 0 : 8;
    const float* X0 = X + (size_t)rowA * K + ks0 + kbA;
    const float* X1 = X + (size_t)(rowA + 16) * K + ks0 + kbA;

    v8f c0 = {}, c1 = {};
    for (int ks = 0; ks < kchunk; ks += 32) {
        __builtin_prefetch(Wp + ks + 128, 0, 1);   // global_prefetch_b8
        v16bf bt, a0, a1;
        const float4* w4 = (const float4*)(Wp + ks);
        f4cvt(bt, 0,  w4[0]); f4cvt(bt, 4,  w4[1]);
        f4cvt(bt, 8,  w4[4]); f4cvt(bt, 12, w4[5]);
        const float4* xa = (const float4*)(X0 + ks);
        const float4* xb = (const float4*)(X0 + ks + 16);
        f4cvt(a0, 0, xa[0]); f4cvt(a0, 4, xa[1]);
        f4cvt(a0, 8, xb[0]); f4cvt(a0, 12, xb[1]);
        const float4* ya = (const float4*)(X1 + ks);
        const float4* yb = (const float4*)(X1 + ks + 16);
        f4cvt(a1, 0, ya[0]); f4cvt(a1, 4, ya[1]);
        f4cvt(a1, 8, yb[0]); f4cvt(a1, 12, yb[1]);
        c0 = __builtin_amdgcn_wmma_f32_16x16x32_bf16(false, a0, false, bt,
                                                     (short)0, c0, false, false);
        c1 = __builtin_amdgcn_wmma_f32_16x16x32_bf16(false, a1, false, bt,
                                                     (short)0, c1, false, false);
    }
    // D layout: VGPR j -> row j (lanes 0-15) / row 8+j (lanes 16-31); col = lane&15
    int cm = (lane < 16) ? 0 : 8;
    int cn = n0 + (lane & 15);
    float* out = ACC + (size_t)blockIdx.y * 32 * N;
    #pragma unroll
    for (int j = 0; j < 8; ++j) {
        out[(size_t)(cm + j) * N + cn]      = c0[j];
        out[(size_t)(cm + 16 + j) * N + cn] = c1[j];
    }
}

// bias + ELU epilogue, deterministic split-K reduction
__global__ __launch_bounds__(256) void pp_elu(const float* __restrict__ ACC,
                                              const float* __restrict__ bias,
                                              float* __restrict__ Y, int N) {
    int i = blockIdx.x * 256 + threadIdx.x;        // 32*N threads
    float v = bias[i % N];
    #pragma unroll
    for (int s = 0; s < KSPLIT; ++s) v += ACC[(size_t)s * 32 * N + i];
    Y[i] = v > 0.f ? v : (__expf(v) - 1.f);
}

// final layer: bias, split into (mean, logvar)
__global__ __launch_bounds__(256) void pp_out(const float* __restrict__ ACC,
                                              const float* __restrict__ bout,
                                              float* __restrict__ out_mean,
                                              float* __restrict__ out_logvar) {
    int i = blockIdx.x * 256 + threadIdx.x;        // 32*256 threads
    int m = i / (2 * CDIM), n = i % (2 * CDIM);
    float v = bout[n];
    #pragma unroll
    for (int s = 0; s < KSPLIT; ++s) v += ACC[(size_t)s * 32 * (2 * CDIM) + i];
    if (n < CDIM) out_mean[m * CDIM + n] = v;
    else          out_logvar[m * CDIM + (n - CDIM)] = v;
}

// ---------------------------------------------------------------------------
// Host launcher
// ---------------------------------------------------------------------------
extern "C" void kernel_launch(void* const* d_in, const int* in_sizes, int n_in,
                              void* d_out, int out_size, void* d_ws, size_t ws_size,
                              hipStream_t stream) {
    const float* h    = (const float*)d_in[0];
    const float* Wq   = (const float*)d_in[1];
    const float* bq   = (const float*)d_in[2];
    const float* Wk   = (const float*)d_in[3];
    const float* bk   = (const float*)d_in[4];
    const float* Wv   = (const float*)d_in[5];
    const float* bv   = (const float*)d_in[6];
    const float* W0   = (const float*)d_in[7];
    const float* b0   = (const float*)d_in[8];
    const float* W1   = (const float*)d_in[9];
    const float* b1   = (const float*)d_in[10];
    const float* W2   = (const float*)d_in[11];
    const float* b2   = (const float*)d_in[12];
    const float* Wout = (const float*)d_in[13];
    const float* bout = (const float*)d_in[14];

    float* out = (float*)d_out;
    float* out_mean   = out;                       //  32*128
    float* out_logvar = out + BB * CDIM;           //  32*128
    float* out_a      = out + 2 * BB * CDIM;       //  32*4096  (also x0)
    float* out_att    = out_a + BB * CP;           //  32*4*512

    // workspace layout (floats)
    float* ws   = (float*)d_ws;
    float* z    = ws;                              // 131072
    float* q    = z    + BB * CP;                  // 131072
    float* qk   = q    + BB * CP;                  // 524288
    float* qb   = qk   + BB * NHEAD * CP;          // 128
    float* sim  = qb   + 128;                      // 65536
    float* attw = sim  + BB * NHEAD * TT;          // 65536
    float* hbar = attw + BB * NHEAD * TT;          // 524288
    float* x1   = hbar + BB * NHEAD * CP;          // 131072
    float* x2   = x1   + BB * FEAT;                // 131072
    float* x3   = x2   + BB * FEAT;                // 131072
    float* acc  = x3   + BB * FEAT;                // KSPLIT*32*4096 = 1048576

    // ---- attention (algebraically fused: 3 streaming passes over h) ----
    mean_kernel   <<<BB * CP / 256, 256, 0, stream>>>(h, z);
    qproj_kernel  <<<BB * CP / 256, 256, 0, stream>>>(z, Wq, bq, q);
    qk_kernel     <<<BB * NHEAD * CP / 256, 256, 0, stream>>>(q, Wk, qk);
    qb_kernel     <<<1, 128, 0, stream>>>(q, bk, qb);
    sim_kernel    <<<BB * TT / 8, 256, 0, stream>>>(h, qk, qb, sim);
    softmax_kernel<<<BB, 128, 0, stream>>>(sim, attw, out_att);
    hbar_kernel   <<<BB * 16, 256, 0, stream>>>(h, attw, hbar);
    aproj_kernel  <<<BB * CP / 256, 256, 0, stream>>>(hbar, Wv, bv, out_a);

    // ---- PostPool: bf16 WMMA GEMMs, split-K=8, deterministic reduce ----
    const int kchunk = FEAT / KSPLIT;              // 512
    pp_gemm<<<dim3(FEAT / 16, KSPLIT), 32, 0, stream>>>(out_a, W0, acc, FEAT, FEAT, kchunk);
    pp_elu <<<BB * FEAT / 256, 256, 0, stream>>>(acc, b0, x1, FEAT);
    pp_gemm<<<dim3(FEAT / 16, KSPLIT), 32, 0, stream>>>(x1, W1, acc, FEAT, FEAT, kchunk);
    pp_elu <<<BB * FEAT / 256, 256, 0, stream>>>(acc, b1, x2, FEAT);
    pp_gemm<<<dim3(FEAT / 16, KSPLIT), 32, 0, stream>>>(x2, W2, acc, FEAT, FEAT, kchunk);
    pp_elu <<<BB * FEAT / 256, 256, 0, stream>>>(acc, b2, x3, FEAT);
    pp_gemm<<<dim3((2 * CDIM) / 16, KSPLIT), 32, 0, stream>>>(x3, Wout, acc, 2 * CDIM, FEAT, kchunk);
    pp_out <<<BB * 2 * CDIM / 256, 256, 0, stream>>>(acc, bout, out_mean, out_logvar);

    (void)in_sizes; (void)n_in; (void)out_size; (void)ws_size;
}
